// MovementPredictionHead_53944789238376
// MI455X (gfx1250) — compile-verified, matrix operand order checked by reading, not run on hardware
//
#include <hip/hip_runtime.h>
#include <hip/hip_bf16.h>
#include <math.h>

#define N_NODE 20000
#define N_EDGE 200000
#define EMBED 256
#define PAIR_D 128
#define HEADS 16
#define HEAD_DIM 16
#define SCALING 0.25f
#define LN_EPS 1e-5f

typedef __attribute__((ext_vector_type(16))) __bf16 v16bf;
typedef __attribute__((ext_vector_type(8)))  __bf16 v8bf;
typedef __attribute__((ext_vector_type(8)))  float  v8f;

// ---------- helpers ----------
__device__ __forceinline__ float wave_reduce_add(float v) {
    v += __shfl_xor(v, 16, 32);
    v += __shfl_xor(v, 8, 32);
    v += __shfl_xor(v, 4, 32);
    v += __shfl_xor(v, 2, 32);
    v += __shfl_xor(v, 1, 32);
    return v;
}

// monotonic float<->uint mapping so atomicMax(uint) == float max
__device__ __forceinline__ unsigned int f32_to_ord(float f) {
    unsigned int u = __float_as_uint(f);
    return (u & 0x80000000u) ? ~u : (u | 0x80000000u);
}
__device__ __forceinline__ float ord_to_f32(unsigned int u) {
    unsigned int b = (u & 0x80000000u) ? (u & 0x7FFFFFFFu) : ~u;
    return __uint_as_float(b);
}
#define ORD_NEG_INF 0x007FFFFFu   // encoding of -inf

// ---------- kernel 1: LayerNorm(query) -> bf16 ----------
__global__ void k_ln_query(const float* __restrict__ q,
                           const float* __restrict__ w,
                           const float* __restrict__ b,
                           __bf16* __restrict__ qn) {
    int wave = threadIdx.x >> 5, lane = threadIdx.x & 31;
    int row = blockIdx.x * 8 + wave;
    if (row >= N_NODE) return;
    const float* r = q + (size_t)row * EMBED;
    float x[8];
    float4 a0 = *(const float4*)(r + lane * 8);
    float4 a1 = *(const float4*)(r + lane * 8 + 4);
    x[0]=a0.x; x[1]=a0.y; x[2]=a0.z; x[3]=a0.w;
    x[4]=a1.x; x[5]=a1.y; x[6]=a1.z; x[7]=a1.w;
    float s = 0.f, s2 = 0.f;
    #pragma unroll
    for (int i = 0; i < 8; ++i) { s += x[i]; s2 += x[i] * x[i]; }
    s  = wave_reduce_add(s);
    s2 = wave_reduce_add(s2);
    float mean = s * (1.0f / EMBED);
    float inv  = rsqrtf(s2 * (1.0f / EMBED) - mean * mean + LN_EPS);
    __bf16* o = qn + (size_t)row * EMBED + lane * 8;
    #pragma unroll
    for (int i = 0; i < 8; ++i) {
        int c = lane * 8 + i;
        o[i] = (__bf16)((x[i] - mean) * inv * w[c] + b[c]);
    }
}

// ---------- kernel 2: convert Wq/Wk/Wv to bf16 ----------
__global__ void k_cvt_w(const float* __restrict__ Wq, const float* __restrict__ Wk,
                        const float* __restrict__ Wv, __bf16* __restrict__ out) {
    int i = blockIdx.x * blockDim.x + threadIdx.x;
    if (i < EMBED * EMBED) {
        out[i]                   = (__bf16)Wq[i];
        out[EMBED * EMBED + i]   = (__bf16)Wk[i];
        out[2 * EMBED * EMBED + i] = (__bf16)Wv[i];
    }
}

// ---------- kernel 3: init accumulators ----------
__global__ void k_init(unsigned int* __restrict__ segmax, float* __restrict__ denom,
                       float* __restrict__ xacc) {
    int i = blockIdx.x * blockDim.x + threadIdx.x;     // 320000 threads
    if (i < N_NODE * HEADS) { segmax[i] = ORD_NEG_INF; denom[i] = 0.f; }
    int total = N_NODE * 3 * EMBED;
    int stride = gridDim.x * blockDim.x;
    for (int j = i; j < total; j += stride) xacc[j] = 0.f;
}

// ---------- kernel 4: QKV projection via WMMA bf16 ----------
// out[n,o] = sum_e qn[n,e] * W[o,e]; one wave per 16x16 tile, K=256 in 8 chunks
__global__ void k_qkv(const __bf16* __restrict__ qn, const __bf16* __restrict__ wmat,
                      float* __restrict__ qkv) {
    int wave = threadIdx.x >> 5, lane = threadIdx.x & 31;
    int tile = blockIdx.x * 8 + wave;                  // 1250*16*3 tiles
    if (tile >= 1250 * 16 * 3) return;
    int mat = tile / (1250 * 16);
    int rem = tile % (1250 * 16);
    int tm = rem / 16, tn = rem % 16;
    int half = lane >> 4;
    const __bf16* arow = qn + (size_t)(tm * 16 + (lane & 15)) * EMBED;
    const __bf16* brow = wmat + (size_t)mat * EMBED * EMBED
                              + (size_t)(tn * 16 + (lane & 15)) * EMBED;
    v8f c = {};
    #pragma unroll
    for (int kk = 0; kk < EMBED; kk += 32) {
        v8bf a0 = *(const v8bf*)(arow + kk + half * 8);
        v8bf a1 = *(const v8bf*)(arow + kk + 16 + half * 8);
        v8bf b0 = *(const v8bf*)(brow + kk + half * 8);
        v8bf b1 = *(const v8bf*)(brow + kk + 16 + half * 8);
        v16bf A = __builtin_shufflevector(a0, a1, 0,1,2,3,4,5,6,7,8,9,10,11,12,13,14,15);
        v16bf B = __builtin_shufflevector(b0, b1, 0,1,2,3,4,5,6,7,8,9,10,11,12,13,14,15);
        c = __builtin_amdgcn_wmma_f32_16x16x32_bf16(false, A, false, B, (short)0, c,
                                                    false, false);
    }
    float* out = qkv + (size_t)mat * N_NODE * EMBED;
    int m0 = tm * 16 + 8 * half;
    int n  = tn * 16 + (lane & 15);
    #pragma unroll
    for (int r = 0; r < 8; ++r) out[(size_t)(m0 + r) * EMBED + n] = c[r];
}

// ---------- kernel 5: pair LN + bias WMMA + q.k logits + segment max ----------
// one wave per 16-edge tile; A = LN(pair) rows (16x128), B = Wb^T (128x16)
__global__ void k_logits(const float* __restrict__ pair, const float* __restrict__ pw,
                         const float* __restrict__ pb, const float* __restrict__ Wb,
                         const float* __restrict__ bb, const int* __restrict__ eidx,
                         const float* __restrict__ qbuf, const float* __restrict__ kbuf,
                         float* __restrict__ logits, unsigned int* __restrict__ segmax) {
    int wave = threadIdx.x >> 5, lane = threadIdx.x & 31;
    int tile = blockIdx.x * 8 + wave;                  // N_EDGE/16 = 12500 tiles
    if (tile >= N_EDGE / 16) return;
    int half = lane >> 4;
    int erow = tile * 16 + (lane & 15);
    const float* prow = pair + (size_t)erow * PAIR_D;

    // this lane owns K = kk*32 + g*16 + half*8 + t  (64 of the 128 row elements)
    float pv[64];
    float s = 0.f, s2 = 0.f;
    #pragma unroll
    for (int kk = 0; kk < 4; ++kk) {
        #pragma unroll
        for (int g = 0; g < 2; ++g) {
            const float* p = prow + kk * 32 + g * 16 + half * 8;
            float4 u0 = *(const float4*)p;
            float4 u1 = *(const float4*)(p + 4);
            float* d = &pv[kk * 16 + g * 8];
            d[0]=u0.x; d[1]=u0.y; d[2]=u0.z; d[3]=u0.w;
            d[4]=u1.x; d[5]=u1.y; d[6]=u1.z; d[7]=u1.w;
            s  += u0.x + u0.y + u0.z + u0.w + u1.x + u1.y + u1.z + u1.w;
            s2 += u0.x*u0.x + u0.y*u0.y + u0.z*u0.z + u0.w*u0.w
                + u1.x*u1.x + u1.y*u1.y + u1.z*u1.z + u1.w*u1.w;
        }
    }
    // combine row halves (rows live on lanes r and r+16)
    s  += __shfl_xor(s, 16, 32);
    s2 += __shfl_xor(s2, 16, 32);
    float mean = s * (1.0f / PAIR_D);
    float inv  = rsqrtf(s2 * (1.0f / PAIR_D) - mean * mean + LN_EPS);

    int hn = lane & 15;                                // head column for B
    const float* wbrow = Wb + (size_t)hn * PAIR_D;
    v8f c = {};
    #pragma unroll
    for (int kk = 0; kk < 4; ++kk) {
        v16bf A, B;
        #pragma unroll
        for (int g = 0; g < 2; ++g) {
            #pragma unroll
            for (int t = 0; t < 8; ++t) {
                int K = kk * 32 + g * 16 + half * 8 + t;
                float xv = (pv[kk * 16 + g * 8 + t] - mean) * inv;
                A[g * 8 + t] = (__bf16)(xv * pw[K] + pb[K]);
                B[g * 8 + t] = (__bf16)wbrow[K];
            }
        }
        c = __builtin_amdgcn_wmma_f32_16x16x32_bf16(false, A, false, B, (short)0, c,
                                                    false, false);
    }

    // finish: c[r] holds bias for (edge = tile*16 + 8*half + r, head = hn)
    #pragma unroll
    for (int r = 0; r < 8; ++r) {
        int e   = tile * 16 + 8 * half + r;
        int src = eidx[2 * e], dst = eidx[2 * e + 1];
        const float* qr = qbuf + (size_t)src * EMBED + hn * HEAD_DIM;
        const float* kr = kbuf + (size_t)dst * EMBED + hn * HEAD_DIM;
        float dot = 0.f;
        #pragma unroll
        for (int d = 0; d < HEAD_DIM; d += 4) {
            float4 a = *(const float4*)(qr + d);
            float4 b = *(const float4*)(kr + d);
            dot += a.x*b.x + a.y*b.y + a.z*b.z + a.w*b.w;
        }
        float lg = dot * SCALING + c[r] + bb[hn];
        logits[(size_t)e * HEADS + hn] = lg;
        atomicMax(&segmax[(size_t)src * HEADS + hn], f32_to_ord(lg));
    }
}

// ---------- kernel 6: exp(logit - segmax) + segment sum ----------
__global__ void k_exp(const int* __restrict__ eidx, const unsigned int* __restrict__ segmax,
                      float* __restrict__ logits, float* __restrict__ denom) {
    int i = blockIdx.x * blockDim.x + threadIdx.x;     // E*16 = 3.2M
    if (i >= N_EDGE * HEADS) return;
    int e = i >> 4, h = i & 15;
    int src = eidx[2 * e];
    float m  = ord_to_f32(segmax[(size_t)src * HEADS + h]);
    float ex = __expf(logits[i] - m);
    logits[i] = ex;                                     // reuse buffer for ex
    atomicAdd(&denom[(size_t)src * HEADS + h], ex);
}

// ---------- kernel 7: weighted scatter of v[dst] into xacc[src][3][256] ----------
__global__ void k_contrib(const int* __restrict__ eidx, const float* __restrict__ ex,
                          const float* __restrict__ denom, const float* __restrict__ ediff,
                          const float* __restrict__ vbuf, float* __restrict__ xacc) {
    int wave = threadIdx.x >> 5, lane = threadIdx.x & 31;
    int e = blockIdx.x * 8 + wave;
    if (e >= N_EDGE) return;
    int src = eidx[2 * e], dst = eidx[2 * e + 1];
    int h = lane >> 1;                                 // lane's 8 cols share one head
    float p  = ex[(size_t)e * HEADS + h] / denom[(size_t)src * HEADS + h];
    float d0 = ediff[e * 3 + 0], d1 = ediff[e * 3 + 1], d2 = ediff[e * 3 + 2];
    const float* vr = vbuf + (size_t)dst * EMBED + lane * 8;
    float4 v0 = *(const float4*)vr;
    float4 v1 = *(const float4*)(vr + 4);
    float vv[8] = { v0.x, v0.y, v0.z, v0.w, v1.x, v1.y, v1.z, v1.w };
    float* xb = xacc + (size_t)src * 3 * EMBED;
    #pragma unroll
    for (int i = 0; i < 8; ++i) {
        float t = p * vv[i];
        int col = lane * 8 + i;
        atomicAdd(xb + col,             t * d0);
        atomicAdd(xb + EMBED + col,     t * d1);
        atomicAdd(xb + 2 * EMBED + col, t * d2);
    }
}

// ---------- kernel 8: final 256->1 dots ----------
__global__ void k_final(const float* __restrict__ xacc,
                        const float* __restrict__ Wf1, const float* __restrict__ bf1,
                        const float* __restrict__ Wf2, const float* __restrict__ bf2,
                        const float* __restrict__ Wf3, const float* __restrict__ bf3,
                        float* __restrict__ out) {
    int wave = threadIdx.x >> 5, lane = threadIdx.x & 31;
    int n = blockIdx.x * 8 + wave;
    if (n >= N_NODE) return;
    const float* xb = xacc + (size_t)n * 3 * EMBED;
    const float* Wf[3] = { Wf1, Wf2, Wf3 };
    const float* bf[3] = { bf1, bf2, bf3 };
    #pragma unroll
    for (int ci = 0; ci < 3; ++ci) {
        float s = 0.f;
        #pragma unroll
        for (int i = 0; i < 8; i += 4) {
            float4 a = *(const float4*)(xb + ci * EMBED + lane * 8 + i);
            float4 w = *(const float4*)(Wf[ci] + lane * 8 + i);
            s += a.x*w.x + a.y*w.y + a.z*w.z + a.w*w.w;
        }
        s = wave_reduce_add(s);
        if (lane == 0) out[n * 3 + ci] = s + bf[ci][0];
    }
}

// ---------- launch ----------
extern "C" void kernel_launch(void* const* d_in, const int* in_sizes, int n_in,
                              void* d_out, int out_size, void* d_ws, size_t ws_size,
                              hipStream_t stream) {
    (void)in_sizes; (void)n_in; (void)out_size; (void)ws_size;
    const float* query = (const float*)d_in[0];
    const int*   eidx  = (const int*)d_in[1];
    const float* ediff = (const float*)d_in[2];
    const float* pair  = (const float*)d_in[3];
    const float* ln_w  = (const float*)d_in[4];
    const float* ln_b  = (const float*)d_in[5];
    const float* pln_w = (const float*)d_in[6];
    const float* pln_b = (const float*)d_in[7];
    const float* Wq    = (const float*)d_in[8];
    const float* Wk    = (const float*)d_in[9];
    const float* Wv    = (const float*)d_in[10];
    const float* Wb    = (const float*)d_in[11];
    const float* bb    = (const float*)d_in[12];
    const float* Wf1   = (const float*)d_in[13];
    const float* bf1   = (const float*)d_in[14];
    const float* Wf2   = (const float*)d_in[15];
    const float* bf2   = (const float*)d_in[16];
    const float* Wf3   = (const float*)d_in[17];
    const float* bf3   = (const float*)d_in[18];

    // workspace layout (all 256B aligned), total ~142 MB
    char* ws = (char*)d_ws;
    __bf16*       qn     = (__bf16*)(ws + 0);               // 20000*256*2  = 10,240,000
    __bf16*       wmat   = (__bf16*)(ws + 10240000);        // 3*256*256*2  =    393,216
    float*        qkv    = (float*) (ws + 10633216);        // 3*20000*256*4= 61,440,000
    float*        logits = (float*) (ws + 72073216);        // 200000*16*4  = 12,800,000
    unsigned int* segmax = (unsigned int*)(ws + 84873216);  // 20000*16*4   =  1,280,000
    float*        denom  = (float*) (ws + 86153216);        // 20000*16*4   =  1,280,000
    float*        xacc   = (float*) (ws + 87433216);        // 20000*768*4  = 61,440,000

    float* qb = qkv;
    float* kb = qkv + (size_t)N_NODE * EMBED;
    float* vb = qkv + (size_t)2 * N_NODE * EMBED;

    k_ln_query<<<2500, 256, 0, stream>>>(query, ln_w, ln_b, qn);
    k_cvt_w  <<<256, 256, 0, stream>>>(Wq, Wk, Wv, wmat);
    k_init   <<<1250, 256, 0, stream>>>(segmax, denom, xacc);
    k_qkv    <<<7500, 256, 0, stream>>>(qn, wmat, qkv);
    k_logits <<<1563, 256, 0, stream>>>(pair, pln_w, pln_b, Wb, bb, eidx,
                                        qb, kb, logits, segmax);
    k_exp    <<<12500, 256, 0, stream>>>(eidx, segmax, logits, denom);
    k_contrib<<<25000, 256, 0, stream>>>(eidx, logits, denom, ediff, vb, xacc);
    k_final  <<<2500, 256, 0, stream>>>(xacc, Wf1, bf1, Wf2, bf2, Wf3, bf3,
                                        (float*)d_out);
}
